// PRMLayer_21002390077821
// MI455X (gfx1250) — matrix operand order repaired and checked by reading, not compiled, not run on hardware
//
#include <hip/hip_runtime.h>
#include <math.h>

typedef __attribute__((ext_vector_type(16))) _Float16 v16h;
typedef __attribute__((ext_vector_type(8)))  float    v8f;

#define BB   8
#define CC   128
#define HH   48
#define NPOS 110592                 // 48^3
#define HW   2304                   // 48*48
#define QSTRIDE 2353                // 2304 + 48 + 1  (flat offset of (i0,i0,i0))
#define CN   (128ull*110592ull)     // elements per batch

// ---------------- init: zero all accumulators (ws is poisoned) ----------------
__global__ void k_init(float* gapsum, unsigned long long* best, float* ssum, float* ssumsq) {
  int t = threadIdx.x;
  for (int i = t; i < BB*CC; i += blockDim.x) gapsum[i] = 0.f;
  if (t < BB) { best[t] = 0ull; ssum[t] = 0.f; ssumsq[t] = 0.f; }
}

// ---------------- pass 1: channel sums -> argmax key + gap partial sums ----------------
__global__ void k_reduce(const float* __restrict__ x, float* __restrict__ gapsum,
                         unsigned long long* __restrict__ best) {
  __shared__ float s_part[CC*8];
  __shared__ unsigned long long s_am[256];
  const int b    = blockIdx.x / 432;
  const int chnk = blockIdx.x % 432;
  const int tid  = threadIdx.x;
  const int wave = tid >> 5, lane = tid & 31;
  const int pos  = chnk*256 + tid;
  const float* xb = x + (size_t)b*CN + pos;

  float acc = 0.f;
  #pragma unroll 4
  for (int c = 0; c < CC; ++c) {
    float v = xb[(size_t)c*NPOS];
    acc += v;
    float g = v;                       // wave-reduce over 32 voxels for gap[b,c]
    g += __shfl_xor(g, 16, 32);
    g += __shfl_xor(g,  8, 32);
    g += __shfl_xor(g,  4, 32);
    g += __shfl_xor(g,  2, 32);
    g += __shfl_xor(g,  1, 32);
    if (lane == 0) s_part[c*8 + wave] = g;
  }
  __syncthreads();
  if (tid < CC) {
    float g = 0.f;
    #pragma unroll
    for (int w = 0; w < 8; ++w) g += s_part[tid*8 + w];
    atomicAdd(&gapsum[b*CC + tid], g);
  }
  // sortable-float key; ~pos in low bits so ties pick the smallest index (jnp.argmax)
  unsigned int u   = __float_as_uint(acc);
  unsigned int key = (u & 0x80000000u) ? ~u : (u | 0x80000000u);
  unsigned long long pk = ((unsigned long long)key << 32)
                        | (unsigned long long)(0xFFFFFFFFu - (unsigned int)pos);
  s_am[tid] = pk;
  __syncthreads();
  for (int s = 128; s > 0; s >>= 1) {
    if (tid < s) { unsigned long long o = s_am[tid + s]; if (o > s_am[tid]) s_am[tid] = o; }
    __syncthreads();
  }
  if (tid == 0) atomicMax(&best[b], s_am[0]);
}

// ---------------- tiny: decode argmax, gather qval, pack WMMA A-matrices ----------------
// A (16x32 f16, MxK) layout per CDNA5 ISA: lane<16 holds M=lane, halves j: K = j (j<8), j+8 (j>=8);
// lane>=16 holds M=lane-16, halves j: K = j+8 (j<8), j+16 (j>=8). Row0=qval, Row1=gap, rest 0.
__global__ void k_prep(const float* __restrict__ x, const unsigned long long* __restrict__ best,
                       float* __restrict__ gapsum, int* __restrict__ qi0,
                       _Float16* __restrict__ apack) {
  __shared__ float s_q[CC], s_g[CC];
  const int b = blockIdx.x, tid = threadIdx.x;
  unsigned long long pk = best[b];
  unsigned int pos = 0xFFFFFFFFu - (unsigned int)(pk & 0xFFFFFFFFull);
  int i0 = (int)(pos / HW);
  float qv = x[(size_t)(b*CC + tid)*NPOS + (size_t)i0*QSTRIDE];
  float gv = gapsum[b*CC + tid] * (1.f/(float)NPOS);
  s_q[tid] = qv; s_g[tid] = gv;
  if (tid == 0) qi0[b] = i0;
  __syncthreads();
  const int ck = tid >> 5, lane = tid & 31, m = lane & 15;
  _Float16* ap = apack + (size_t)(((b*4 + ck)*32 + lane) * 16);
  #pragma unroll
  for (int j = 0; j < 16; ++j) {
    int kl = j + (j >= 8 ? 8 : 0) + (lane >= 16 ? 8 : 0);
    int kk = ck*32 + kl;
    float val = (m == 0) ? s_q[kk] : (m == 1) ? s_g[kk] : 0.f;
    ap[j] = (_Float16)val;
  }
}

// ---------------- pass 2: WMMA [qval;gap] @ X, analytic dw, sim + stats ----------------
__global__ void k_sim(const float* __restrict__ x, const _Float16* __restrict__ apackp,
                      const int* __restrict__ qi0,
                      const float* __restrict__ theta, const float* __restrict__ scale,
                      const float* __restrict__ onep, const float* __restrict__ zerop,
                      float* __restrict__ simb, float* __restrict__ ssum,
                      float* __restrict__ ssumsq) {
  const int b   = blockIdx.x / 864;
  const int tch = blockIdx.x % 864;
  const int tid = threadIdx.x, wave = tid >> 5, lane = tid & 31;
  const int pos_base = tch*128 + wave*16;       // 16 voxels per wave-tile
  const int nloc  = lane & 15;                  // B-matrix N = lane%16
  const int khalf = lane & 16;                  // lanes>=16 carry K+8
  const v16h* ap = (const v16h*)apackp;
  v16h a[4];
  #pragma unroll
  for (int k = 0; k < 4; ++k) a[k] = ap[(b*4 + k)*32 + lane];

  const float* xb = x + (size_t)b*CN + (size_t)(pos_base + nloc);
  v8f acc = {};
  #pragma unroll
  for (int k = 0; k < 4; ++k) {
    v16h bv;                                    // B 32x16 f16: VGPR v -> K=khalf/2+2v(+8 if v>=4),+1
    #pragma unroll
    for (int v = 0; v < 8; ++v) {
      const float* p = xb + (size_t)(k*32 + (khalf >> 1) + 2*v + (v >= 4 ? 8 : 0)) * NPOS;
      bv[2*v]   = (_Float16)p[0];
      bv[2*v+1] = (_Float16)p[NPOS];
    }
    acc = __builtin_amdgcn_wmma_f32_16x16x32_f16(false, a[k], false, bv,
                                                 (short)0, acc, false, false);
  }

  float sval = 0.f;
  if (lane < 16) {                              // D: acc[0]=M0 (sim_max), acc[1]=M1 (sim_gap)
    float simmax = acc[0];
    float simgap = acc[1];
    int pos = pos_base + lane;
    int i0 = pos / HW; int r = pos - i0*HW; int i1 = r / HH; int i2 = r - i1*HH;
    int q = qi0[b];
    int d0 = i0 - q; d0 = d0 < 0 ? -d0 : d0;
    int d1 = i1 - q; d1 = d1 < 0 ? -d1 : d1;
    int d2 = i2 - q; d2 = d2 < 0 ? -d2 : d2;
    float inv = 1.f / scale[0];
    float z0 = (float)d0 * theta[0] * inv;
    float z1 = (float)d1 * theta[1] * inv;
    float z2 = (float)d2 * theta[2] * inv;
    float e  = expf(-0.5f*z0*z0) + expf(-0.5f*z1*z1) + expf(-0.5f*z2*z2);
    float dw = e * (0.3989422804014327f * inv * (1.f/3.f));   // mean of 3 gaussians
    sval = simmax * dw * zerop[0] + simgap * onep[0];
    simb[(size_t)b*NPOS + pos] = sval;
  }
  __shared__ float sA[256], sB2[256];
  sA[tid] = sval; sB2[tid] = sval*sval;
  __syncthreads();
  for (int s = 128; s > 0; s >>= 1) {
    if (tid < s) { sA[tid] += sA[tid+s]; sB2[tid] += sB2[tid+s]; }
    __syncthreads();
  }
  if (tid == 0) { atomicAdd(&ssum[b], sA[0]); atomicAdd(&ssumsq[b], sB2[0]); }
}

// ---------------- pass 3: streaming gate + multiply ----------------
__global__ void k_out(const float* __restrict__ x, const float* __restrict__ simb,
                      const float* __restrict__ ssum, const float* __restrict__ ssumsq,
                      const float* __restrict__ weight, const float* __restrict__ bias,
                      float* __restrict__ out) {
  size_t i    = (size_t)blockIdx.x * blockDim.x + threadIdx.x;
  size_t base = i * 4;
  int b      = (int)(base / CN);
  size_t rem = base - (size_t)b * CN;
  int pos    = (int)(rem % (size_t)NPOS);
  float4 xv = ((const float4*)x)[i];
  float4 sv = ((const float4*)simb)[(size_t)b*(NPOS/4) + (pos >> 2)];
  float s1   = ssum[b];
  float mean = s1 * (1.f/(float)NPOS);
  float var  = (ssumsq[b] - s1*mean) * (1.f/(float)(NPOS-1));   // ddof=1
  float invstd = 1.f / (sqrtf(var) + 1e-5f);
  float A  = invstd * weight[0];
  float Bc = bias[0] - mean * A;
  float g0 = 1.f/(1.f + __expf(-(sv.x*A + Bc)));
  float g1 = 1.f/(1.f + __expf(-(sv.y*A + Bc)));
  float g2 = 1.f/(1.f + __expf(-(sv.z*A + Bc)));
  float g3 = 1.f/(1.f + __expf(-(sv.w*A + Bc)));
  ((float4*)out)[i] = make_float4(xv.x*g0, xv.y*g1, xv.z*g2, xv.w*g3);
}

extern "C" void kernel_launch(void* const* d_in, const int* in_sizes, int n_in,
                              void* d_out, int out_size, void* d_ws, size_t ws_size,
                              hipStream_t stream) {
  const float* x      = (const float*)d_in[0];
  const float* weight = (const float*)d_in[1];
  const float* bias   = (const float*)d_in[2];
  const float* one    = (const float*)d_in[3];
  const float* zero   = (const float*)d_in[4];
  const float* theta  = (const float*)d_in[5];
  const float* scale  = (const float*)d_in[6];
  float* out = (float*)d_out;

  char* ws = (char*)d_ws;
  float*              gapsum = (float*)(ws + 0);        // 1024 f = 4096 B
  unsigned long long* best   = (unsigned long long*)(ws + 4096);   // 64 B
  float*              ssum   = (float*)(ws + 4160);     // 8 f
  float*              ssumsq = (float*)(ws + 4192);     // 8 f
  int*                qi0    = (int*)(ws + 4224);       // 8 i
  _Float16*           apack  = (_Float16*)(ws + 4288);  // 8*4*32*16*2 = 32768 B (32B aligned)
  float*              simb   = (float*)(ws + 37120);    // 8*110592 f = 3538944 B (16B aligned)

  k_init  <<<1, 256, 0, stream>>>(gapsum, best, ssum, ssumsq);
  k_reduce<<<BB*432, 256, 0, stream>>>(x, gapsum, best);
  k_prep  <<<BB, 128, 0, stream>>>(x, best, gapsum, qi0, apack);
  k_sim   <<<BB*864, 256, 0, stream>>>(x, apack, qi0, theta, scale, one, zero,
                                       simb, ssum, ssumsq);
  k_out   <<<(unsigned)((BB*CN/4)/256), 256, 0, stream>>>(x, simb, ssum, ssumsq,
                                                          weight, bias, out);
}
